// LayoutModel_63342177681920
// MI455X (gfx1250) — matrix operand order repaired and checked by reading, not compiled, not run on hardware
//
#include <hip/hip_runtime.h>
#include <hip/hip_bf16.h>
#include <stdint.h>

#define Cc   32
#define NCc  2000
#define NN   10000
#define EE   30000
#define OPE  4
#define NF   140
#define CF   18
#define GD   128
#define NL   4

typedef __attribute__((ext_vector_type(16))) __bf16 v16bf;
typedef __attribute__((ext_vector_type(8)))  float  v8f;

__device__ __forceinline__ unsigned short f2bf(float f) {
  unsigned int x = __float_as_uint(f);
  x += 0x7FFFu + ((x >> 16) & 1u);           // round-to-nearest-even
  return (unsigned short)(x >> 16);
}
__device__ __forceinline__ float bf2f(unsigned short h) {
  return __uint_as_float(((unsigned int)h) << 16);
}

union Frag {
  uint4    q[2];
  uint32_t u[8];
  v16bf    v;
};

// ---------------------------------------------------------------------------
// Pre-swizzle conv_Wl/conv_Wr (fp32, row-major KxN) into per-lane WMMA
// B-fragment layout (bf16 pairs packed in dwords).
// Group g = layer*64 + mat*32 + slab*8 + ntile; 256 dwords per group.
// B layout (16-bit 32x16): VGPR v packs K = 8*half + 2*(v&3) + 16*(v>>2) and +1
// at column N = ntile*16 + lane%16.
// ---------------------------------------------------------------------------
__global__ void __launch_bounds__(256) prep_w_kernel(
    const float* __restrict__ Wl, const float* __restrict__ Wr,
    uint32_t* __restrict__ wf)
{
  const int g    = blockIdx.x;            // 0..255
  const int t    = g & 7;                 // n-tile
  const int s    = (g >> 3) & 3;          // k-slab
  const int mat  = (g >> 5) & 1;          // 0 = Wl, 1 = Wr
  const int l    = g >> 6;                // layer
  const int lane = threadIdx.x >> 3;
  const int v    = threadIdx.x & 7;
  const int hf   = lane >> 4;
  const int n    = t * 16 + (lane & 15);
  const int k    = s * 32 + 8 * hf + 2 * (v & 3) + 16 * (v >> 2);
  const float* W = (mat == 0 ? Wl : Wr) + (size_t)l * GD * GD;
  uint32_t lo = f2bf(W[(size_t)k * GD + n]);
  uint32_t hi = f2bf(W[(size_t)(k + 1) * GD + n]);
  wf[(size_t)g * 256 + threadIdx.x] = lo | (hi << 16);
}

// base[n,f] = feat[n]@W[0:140] + emb[x_op[n]]@W[158:162] + b + (-2)*colsum(W[140:158])
__global__ void __launch_bounds__(128) base_kernel(
    const float* __restrict__ x_feat, const int* __restrict__ x_op,
    const float* __restrict__ emb, const float* __restrict__ Wlin,
    const float* __restrict__ blin, float* __restrict__ base)
{
  const int n = blockIdx.x, f = threadIdx.x;
  float s = blin[f];
  const float* xr = x_feat + (size_t)n * NF;
  for (int k = 0; k < NF; ++k)  s += xr[k] * Wlin[(size_t)k * GD + f];
  const int op = x_op[n];
  for (int k = 0; k < OPE; ++k) s += emb[op * OPE + k] * Wlin[(size_t)(NF + CF + k) * GD + f];
  for (int k = 0; k < CF; ++k)  s += -2.0f * Wlin[(size_t)(NF + k) * GD + f];
  base[(size_t)n * GD + f] = s;
}

__global__ void __launch_bounds__(128) bcast_kernel(
    const float* __restrict__ base, unsigned short* __restrict__ x0)
{
  const int n = blockIdx.x, c = blockIdx.y, f = threadIdx.x;
  x0[((size_t)c * NN + n) * GD + f] = f2bf(base[(size_t)n * GD + f]);
}

// Patch config rows: x0[c, ids[j], :] = bf16(base[row] + (cfg[c,j]+2)@Wc)
// ids is sorted; only the last duplicate writes ("last wins").
__global__ void __launch_bounds__(128) cfg_kernel(
    const float* __restrict__ xcfg, const int* __restrict__ ids,
    const float* __restrict__ Wlin, const float* __restrict__ base,
    unsigned short* __restrict__ x0)
{
  const int j = blockIdx.x, c = blockIdx.y, f = threadIdx.x;
  const int row = ids[j];
  if (j + 1 < NCc && ids[j + 1] == row) return;
  float s = base[(size_t)row * GD + f];
  const float* cv = xcfg + ((size_t)c * NCc + j) * CF;
  for (int k = 0; k < CF; ++k)
    s += (cv[k] + 2.0f) * Wlin[(size_t)(NF + k) * GD + f];
  x0[((size_t)c * NN + row) * GD + f] = f2bf(s);
}

__global__ void deg_zero_kernel(float* __restrict__ deg) {
  const int i = blockIdx.x * blockDim.x + threadIdx.x;
  if (i < NN) deg[i] = 0.f;
}
__global__ void deg_add_kernel(const int* __restrict__ dst, float* __restrict__ deg) {
  const int e = blockIdx.x * blockDim.x + threadIdx.x;
  if (e < EE) atomicAdd(&deg[dst[e]], 1.0f);
}

__global__ void zero4_kernel(float4* __restrict__ p, int n4) {
  const int i = blockIdx.x * blockDim.x + threadIdx.x;
  if (i < n4) p[i] = make_float4(0.f, 0.f, 0.f, 0.f);
}

__global__ void __launch_bounds__(128) scatter_kernel(
    const unsigned short* __restrict__ x, const int* __restrict__ src,
    const int* __restrict__ dst, float* __restrict__ agg)
{
  const int e = blockIdx.x, c = blockIdx.y, f = threadIdx.x;
  const int si = src[e], di = dst[e];
  const float v = bf2f(x[((size_t)c * NN + si) * GD + f]);
  atomicAdd(&agg[((size_t)c * NN + di) * GD + f], v);
}

__global__ void __launch_bounds__(128) finalize_kernel(
    const float* __restrict__ agg, const float* __restrict__ deg,
    unsigned short* __restrict__ aggb)
{
  const int n = blockIdx.x, c = blockIdx.y, f = threadIdx.x;
  const float inv = 1.0f / fmaxf(deg[n], 1.0f);
  const size_t idx = ((size_t)c * NN + n) * GD + f;
  aggb[idx] = f2bf(agg[idx] * inv);
}

// ---------------------------------------------------------------------------
// x_next = relu(aggb @ Wl[l] + xcur @ Wr[l] + b[l])   per config
// block = 8 waves; wave w handles N-tile w of one 16-row M-tile.
// 8 chained v_wmma_f32_16x16x32_bf16 per wave.
// A layout (16-bit 16x32, per ISA): lane row = lane%16, half = lane/16;
//   dwords: slab*16 + 4*half + {0..3}  and  slab*16 + 8 + 4*half + {0..3}.
// ---------------------------------------------------------------------------
__global__ void __launch_bounds__(256) gemm_layer_kernel(
    const unsigned short* __restrict__ aggb,
    const unsigned short* __restrict__ xcur,
    const uint32_t* __restrict__ wf,
    const float* __restrict__ convb,
    unsigned short* __restrict__ xnext,
    int layer)
{
  const int mt   = blockIdx.x;            // 0..624 (10000/16)
  const int c    = blockIdx.y;            // 0..31
  const int wave = threadIdx.x >> 5;      // n-tile 0..7
  const int lane = threadIdx.x & 31;
  const int hf   = lane >> 4;
  const int mrow = lane & 15;
  const int mBase = mt * 16;

  const size_t rowA = ((size_t)c * NN + (mBase + mrow)) * GD;
  const uint32_t* aggdw = (const uint32_t*)(aggb + rowA);
  const uint32_t* xdw   = (const uint32_t*)(xcur + rowA);

  v8f acc = {};

  #pragma unroll
  for (int s = 0; s < 4; ++s) {
    Frag a, b;
    const int bd = s * 16 + 4 * hf;
    a.q[0] = *(const uint4*)(aggdw + bd);
    a.q[1] = *(const uint4*)(aggdw + bd + 8);
    const uint32_t* bp = wf + ((((size_t)layer * 2 + 0) * 4 + s) * 8 + wave) * 256 + lane * 8;
    b.q[0] = *(const uint4*)(bp);
    b.q[1] = *(const uint4*)(bp + 4);
    acc = __builtin_amdgcn_wmma_f32_16x16x32_bf16(false, a.v, false, b.v,
                                                  (short)0, acc, false, false);
  }
  #pragma unroll
  for (int s = 0; s < 4; ++s) {
    Frag a, b;
    const int bd = s * 16 + 4 * hf;
    a.q[0] = *(const uint4*)(xdw + bd);
    a.q[1] = *(const uint4*)(xdw + bd + 8);
    const uint32_t* bp = wf + ((((size_t)layer * 2 + 1) * 4 + s) * 8 + wave) * 256 + lane * 8;
    b.q[0] = *(const uint4*)(bp);
    b.q[1] = *(const uint4*)(bp + 4);
    acc = __builtin_amdgcn_wmma_f32_16x16x32_bf16(false, a.v, false, b.v,
                                                  (short)0, acc, false, false);
  }

  // D layout: lane -> N = lane%16; VGPR r -> M = r + 8*(lane/16)
  const int ncol = wave * 16 + mrow;
  const float bias = convb[layer * GD + ncol];
  #pragma unroll
  for (int r = 0; r < 8; ++r) {
    float v = acc[r] + bias;
    v = v > 0.f ? v : 0.f;
    const int m = mBase + r + 8 * hf;
    xnext[((size_t)c * NN + m) * GD + ncol] = f2bf(v);
  }
}

__global__ void __launch_bounds__(128) pool_kernel(
    const unsigned short* __restrict__ x, float* __restrict__ pooled)
{
  const int c = blockIdx.x, f = threadIdx.x;
  float s = 0.f;
  const unsigned short* p = x + (size_t)c * NN * GD + f;
  for (int n = 0; n < NN; ++n) s += bf2f(p[(size_t)n * GD]);
  pooled[c * GD + f] = s * (1.0f / (float)NN);
}

__global__ void __launch_bounds__(128) mlp_kernel(
    const float* __restrict__ pooled,
    const float* __restrict__ Wd1, const float* __restrict__ bd1,
    const float* __restrict__ Wd2, const float* __restrict__ bd2,
    const float* __restrict__ Wd3, const float* __restrict__ bd3,
    float* __restrict__ out)
{
  __shared__ float Abuf[Cc * GD];
  __shared__ float Bbuf[Cc * GD];
  const int t = threadIdx.x;
  for (int i = 0; i < Cc; ++i) Abuf[i * GD + t] = pooled[i * GD + t];
  __syncthreads();
  for (int i = 0; i < Cc; ++i) {
    float s = bd1[t];
    for (int k = 0; k < GD; ++k) s += Abuf[i * GD + k] * Wd1[(size_t)k * GD + t];
    Bbuf[i * GD + t] = fmaxf(s, 0.f);
  }
  __syncthreads();
  for (int i = 0; i < Cc; ++i) {
    float s = bd2[t];
    for (int k = 0; k < GD; ++k) s += Bbuf[i * GD + k] * Wd2[(size_t)k * GD + t];
    Abuf[i * GD + t] = fmaxf(s, 0.f);
  }
  __syncthreads();
  if (t < Cc) {
    float s = bd3[0];
    for (int k = 0; k < GD; ++k) s += Abuf[t * GD + k] * Wd3[k];
    out[t] = s;
  }
}

extern "C" void kernel_launch(void* const* d_in, const int* in_sizes, int n_in,
                              void* d_out, int out_size, void* d_ws, size_t ws_size,
                              hipStream_t stream)
{
  const float* x_node_cfg = (const float*)d_in[0];
  const float* x_feat     = (const float*)d_in[1];
  const int*   x_op       = (const int*)d_in[2];
  const int*   edge_index = (const int*)d_in[3];
  const int*   src        = edge_index;       // edge_index[0]
  const int*   dst        = edge_index + EE;  // edge_index[1]
  const int*   ids        = (const int*)d_in[4];
  const float* emb        = (const float*)d_in[5];
  const float* Wlin       = (const float*)d_in[6];
  const float* blin       = (const float*)d_in[7];
  const float* convWl     = (const float*)d_in[8];
  const float* convWr     = (const float*)d_in[9];
  const float* convb      = (const float*)d_in[10];
  const float* Wd1 = (const float*)d_in[11]; const float* bd1 = (const float*)d_in[12];
  const float* Wd2 = (const float*)d_in[13]; const float* bd2 = (const float*)d_in[14];
  const float* Wd3 = (const float*)d_in[15]; const float* bd3 = (const float*)d_in[16];
  float* out = (float*)d_out;
  (void)in_sizes; (void)n_in; (void)out_size; (void)ws_size;

  char* ws = (char*)d_ws;
  size_t off = 0;
  auto carve = [&](size_t bytes) -> char* {
    char* p = ws + off;
    off = (off + bytes + 255) & ~(size_t)255;
    return p;
  };
  uint32_t*       wf     = (uint32_t*)      carve((size_t)256 * 256 * 4);      // 256 KB
  float*          base   = (float*)         carve((size_t)NN * GD * 4);        // 5.1 MB
  float*          deg    = (float*)         carve((size_t)NN * 4);
  float*          pooled = (float*)         carve((size_t)Cc * GD * 4);
  unsigned short* xA     = (unsigned short*)carve((size_t)Cc * NN * GD * 2);   // 82 MB
  unsigned short* xB     = (unsigned short*)carve((size_t)Cc * NN * GD * 2);   // 82 MB
  float*          aggf   = (float*)         carve((size_t)Cc * NN * GD * 4);   // 164 MB
  unsigned short* aggb   = (unsigned short*)carve((size_t)Cc * NN * GD * 2);   // 82 MB

  prep_w_kernel<<<256, 256, 0, stream>>>(convWl, convWr, wf);
  base_kernel<<<NN, 128, 0, stream>>>(x_feat, x_op, emb, Wlin, blin, base);
  bcast_kernel<<<dim3(NN, Cc), 128, 0, stream>>>(base, xA);
  cfg_kernel<<<dim3(NCc, Cc), 128, 0, stream>>>(x_node_cfg, ids, Wlin, base, xA);
  deg_zero_kernel<<<(NN + 255) / 256, 256, 0, stream>>>(deg);
  deg_add_kernel<<<(EE + 255) / 256, 256, 0, stream>>>(dst, deg);

  unsigned short* xcur = xA;
  unsigned short* xnext = xB;
  const int n4 = Cc * NN * GD / 4;
  for (int l = 0; l < NL; ++l) {
    zero4_kernel<<<(n4 + 255) / 256, 256, 0, stream>>>((float4*)aggf, n4);
    scatter_kernel<<<dim3(EE, Cc), 128, 0, stream>>>(xcur, src, dst, aggf);
    finalize_kernel<<<dim3(NN, Cc), 128, 0, stream>>>(aggf, deg, aggb);
    gemm_layer_kernel<<<dim3(NN / 16, Cc), 256, 0, stream>>>(aggb, xcur, wf, convb, xnext, l);
    unsigned short* tmp = xcur; xcur = xnext; xnext = tmp;
  }

  pool_kernel<<<Cc, 128, 0, stream>>>(xcur, pooled);
  mlp_kernel<<<1, 128, 0, stream>>>(pooled, Wd1, bd1, Wd2, bd2, Wd3, bd3, out);
}
